// SelfAttention_82300163326684
// MI455X (gfx1250) — compile-verified, matrix operand order checked by reading, not run on hardware
//
#include <hip/hip_runtime.h>
#include <hip/hip_bf16.h>
#include <math.h>

// ---------------------------------------------------------------- constants
#define CIN   256          // input channels C
#define DQ    32           // q/k head dim C/8
#define NPIX  4096         // H*W
#define NB    4            // batch

// ---------------------------------------------------------------- types
typedef __attribute__((ext_vector_type(16))) __bf16 v16bf;
typedef __attribute__((ext_vector_type(8)))  float  v8f;

union BFVec {
    v16bf v;
    unsigned short us[16];
    uint4 u4[2];
};

// round-to-nearest-even f32 -> bf16 bits
__device__ __forceinline__ unsigned short f2bf(float f) {
    unsigned u = __float_as_uint(f);
    u += 0x7FFFu + ((u >> 16) & 1u);
    return (unsigned short)(u >> 16);
}

// swap values between the two 16-lane halves of the wave (lane ^= 16)
__device__ __forceinline__ float xhalf(float v) {
#if __has_builtin(__builtin_amdgcn_permlanex16)
    int i = __builtin_amdgcn_permlanex16(__float_as_int(v), __float_as_int(v),
                                         0x76543210, 0xfedcba98, false, false);
    return __int_as_float(i);
#else
    return __shfl_xor(v, 16, 32);
#endif
}

// =====================================================================
// Projection: Y[o,n] = bias[o] + sum_c W[o,c] * X[c,n], 32 output rows per
// block-row-tile. TRANS=1 -> store Y^T as [n][32] bf16 (qt/kt layout),
// TRANS=0 -> store Y as [o][N] bf16 (v layout).
// =====================================================================
template<bool TRANS>
__global__ __launch_bounds__(256)
void proj_kernel(const float* __restrict__ W, const float* __restrict__ bias,
                 const float* __restrict__ x, unsigned short* __restrict__ out)
{
    __shared__ float wl[32 * CIN];               // 32 KB weight tile
    const int b     = blockIdx.z;
    const int j     = blockIdx.x * 256 + threadIdx.x;
    const int rbase = blockIdx.y * 32;

    const float4* Wg  = (const float4*)(W + (size_t)rbase * CIN);
    float4*       wl4 = (float4*)wl;
    #pragma unroll
    for (int idx = threadIdx.x; idx < 32 * (CIN / 4); idx += 256)
        wl4[idx] = Wg[idx];
    __syncthreads();

    float acc[32];
    #pragma unroll
    for (int r = 0; r < 32; ++r) acc[r] = bias[rbase + r];

    const float* xb = x + (size_t)b * CIN * NPIX + j;
    for (int c0 = 0; c0 < CIN; c0 += 4) {
        float x0 = xb[(c0 + 0) * NPIX];
        float x1 = xb[(c0 + 1) * NPIX];
        float x2 = xb[(c0 + 2) * NPIX];
        float x3 = xb[(c0 + 3) * NPIX];
        #pragma unroll
        for (int r = 0; r < 32; ++r) {
            float4 w4 = wl4[r * (CIN / 4) + (c0 >> 2)];
            acc[r] = fmaf(w4.x, x0, fmaf(w4.y, x1, fmaf(w4.z, x2, fmaf(w4.w, x3, acc[r]))));
        }
    }

    if (TRANS) {
        unsigned int packed[16];
        #pragma unroll
        for (int i = 0; i < 16; ++i)
            packed[i] = (unsigned)f2bf(acc[2 * i]) | ((unsigned)f2bf(acc[2 * i + 1]) << 16);
        uint4* dst = (uint4*)(out + ((size_t)b * NPIX + j) * DQ);
        #pragma unroll
        for (int i = 0; i < 4; ++i)
            dst[i] = ((uint4*)packed)[i];
    } else {
        unsigned short* dst = out + (size_t)b * CIN * NPIX;
        #pragma unroll
        for (int r = 0; r < 32; ++r)
            dst[(size_t)(rbase + r) * NPIX + j] = f2bf(acc[r]);
    }
}

// =====================================================================
// Flash attention, transposed-score formulation. Block = 16 query rows (i),
// 2 waves; wave w owns output channels [128w, 128w+128) = 8 D-tiles.
//   S^T tile : A = K rows (j), B = Q rows (i)  ->  D[j][i]
//   The S^T D-layout IS the B-operand layout for the output GEMM:
//   O^T tile : A = V rows (c), B = P^T (k=j, n=i) -> D[c][i], accumulated.
// Softmax over j is lane-local + one xor-16 combine; m/l/scale are
// per-lane (per-i) scalars. No LDS, no cross-lane P movement.
// =====================================================================
__global__ __launch_bounds__(64)
void attn_kernel(const unsigned short* __restrict__ qt,   // [B][N][32] bf16
                 const unsigned short* __restrict__ kt,   // [B][N][32] bf16
                 const unsigned short* __restrict__ vb,   // [B][C][N] bf16
                 const float* __restrict__ x,
                 const float* __restrict__ gamma,
                 float* __restrict__ out)
{
    const int b     = blockIdx.y;
    const int ibase = blockIdx.x * 16;
    const int wave  = threadIdx.x >> 5;
    const int lane  = threadIdx.x & 31;
    const int hn    = lane & 15;                 // A-row / D-col index
    const int h     = lane >> 4;                 // half-wave select
    const int cw    = wave * 128;                // this wave's channel base

    const unsigned short* qtb = qt + (size_t)b * NPIX * DQ;
    const unsigned short* ktb = kt + (size_t)b * NPIX * DQ;
    const unsigned short* vbb = vb + (size_t)b * CIN * NPIX;

    // ---- Q as resident B operand: lane hn holds column i = ibase+hn
    BFVec qb;
    {
        const unsigned short* rowp = qtb + (size_t)(ibase + hn) * DQ;
        qb.u4[0] = *(const uint4*)(rowp + h * 8);
        qb.u4[1] = *(const uint4*)(rowp + 16 + h * 8);
    }

    const v8f zero8 = {0.f, 0.f, 0.f, 0.f, 0.f, 0.f, 0.f, 0.f};
    v8f acc[8];
    #pragma unroll
    for (int t = 0; t < 8; ++t) acc[t] = zero8;

    float mrow = -__builtin_inff();              // running max for row i = ibase+hn
    float lrow = 0.f;                            // running sum

    #pragma unroll 1
    for (int j0 = 0; j0 < NPIX; j0 += 64) {
        // ---- 4 S^T tiles: A = K rows j0+16t+hn, B = Q
        v8f st[4];
        #pragma unroll
        for (int t = 0; t < 4; ++t) {
            const unsigned short* r0 = ktb + (size_t)(j0 + 16 * t + hn) * DQ;
            BFVec ka;
            ka.u4[0] = *(const uint4*)(r0 + h * 8);
            ka.u4[1] = *(const uint4*)(r0 + 16 + h * 8);
            st[t] = __builtin_amdgcn_wmma_f32_16x16x32_bf16(false, ka.v, false, qb.v,
                                                            (short)0, zero8, false, false);
        }

        // ---- online softmax for row i = ibase+hn (lane-local over 32 j's + xor16)
        float mx = st[0][0];
        #pragma unroll
        for (int t = 0; t < 4; ++t)
            #pragma unroll
            for (int r = 0; r < 8; ++r) mx = fmaxf(mx, st[t][r]);
        mx = fmaxf(mx, xhalf(mx));

        float mnew = fmaxf(mrow, mx);
        float sc   = __expf(mrow - mnew);

        float p[4][8];
        float rs = 0.f;
        #pragma unroll
        for (int t = 0; t < 4; ++t)
            #pragma unroll
            for (int r = 0; r < 8; ++r) {
                p[t][r] = __expf(st[t][r] - mnew);
                rs += p[t][r];
            }
        rs += xhalf(rs);
        lrow = lrow * sc + rs;
        mrow = mnew;

        // rescale running accumulators by lane-scalar sc
        #pragma unroll
        for (int t = 0; t < 8; ++t)
            #pragma unroll
            for (int r = 0; r < 8; ++r) acc[t][r] *= sc;

        // ---- pack P^T B-operands directly from S^T D-layout (lane-local)
        BFVec pb0, pb1;
        #pragma unroll
        for (int e = 0; e < 8; ++e) {
            pb0.us[e]     = f2bf(p[0][e]);   // k = 8h+e        (j0 .. j0+15)
            pb0.us[e + 8] = f2bf(p[1][e]);   // k = 16+8h+e-8   (j0+16 .. j0+31)
            pb1.us[e]     = f2bf(p[2][e]);
            pb1.us[e + 8] = f2bf(p[3][e]);
        }

        // ---- O^T accumulate: A = V rows c, two chained WMMAs per c-tile
        #pragma unroll
        for (int t = 0; t < 8; ++t) {
            const unsigned short* vr = vbb + (size_t)(cw + t * 16 + hn) * NPIX + j0;
            BFVec va0, va1;
            va0.u4[0] = *(const uint4*)(vr + h * 8);
            va0.u4[1] = *(const uint4*)(vr + 16 + h * 8);
            va1.u4[0] = *(const uint4*)(vr + 32 + h * 8);
            va1.u4[1] = *(const uint4*)(vr + 48 + h * 8);
            acc[t] = __builtin_amdgcn_wmma_f32_16x16x32_bf16(false, va0.v, false, pb0.v,
                                                             (short)0, acc[t], false, false);
            acc[t] = __builtin_amdgcn_wmma_f32_16x16x32_bf16(false, va1.v, false, pb1.v,
                                                             (short)0, acc[t], false, false);
        }
    }

    // ---- epilogue: out[c][i] = gamma * (O / l) + x ; l is lane-scalar
    const float g   = gamma[0];
    const float inv = 1.0f / lrow;

    const float* xb = x   + (size_t)b * CIN * NPIX;
    float*       ob = out + (size_t)b * CIN * NPIX;
    #pragma unroll
    for (int t = 0; t < 8; ++t) {
        #pragma unroll
        for (int r = 0; r < 8; ++r) {
            int c = cw + t * 16 + r + 8 * h;     // D row -> channel
            size_t idx = (size_t)c * NPIX + (ibase + hn);
            ob[idx] = g * acc[t][r] * inv + xb[idx];
        }
    }
}

// =====================================================================
extern "C" void kernel_launch(void* const* d_in, const int* in_sizes, int n_in,
                              void* d_out, int out_size, void* d_ws, size_t ws_size,
                              hipStream_t stream)
{
    const float* x     = (const float*)d_in[0];
    const float* wq    = (const float*)d_in[1];
    const float* bq    = (const float*)d_in[2];
    const float* wk    = (const float*)d_in[3];
    const float* bk    = (const float*)d_in[4];
    const float* wv    = (const float*)d_in[5];
    const float* bv    = (const float*)d_in[6];
    const float* gamma = (const float*)d_in[7];
    float* out = (float*)d_out;

    // workspace: qt [B][N][32] bf16 | kt [B][N][32] bf16 | v [B][C][N] bf16
    unsigned short* qt = (unsigned short*)d_ws;
    unsigned short* kt = qt + (size_t)NB * NPIX * DQ;
    unsigned short* vw = kt + (size_t)NB * NPIX * DQ;

    proj_kernel<true ><<<dim3(NPIX / 256, 1, NB), dim3(256), 0, stream>>>(wq, bq, x, qt);
    proj_kernel<true ><<<dim3(NPIX / 256, 1, NB), dim3(256), 0, stream>>>(wk, bk, x, kt);
    proj_kernel<false><<<dim3(NPIX / 256, CIN / 32, NB), dim3(256), 0, stream>>>(wv, bv, x, vw);

    attn_kernel<<<dim3(NPIX / 16, NB), dim3(64), 0, stream>>>(qt, kt, vw, x, gamma, out);
}
